// GNN_model_2972117369040
// MI455X (gfx1250) — matrix operand order repaired and checked by reading, not compiled, not run on hardware
//
#include <hip/hip_runtime.h>

#define NNODES 131072
#define NEDGES 262144
#define DIM    128
#define ETOT   (NEDGES + NNODES)   // edges + self loops
#define BN_EPS 1e-5f

typedef __attribute__((ext_vector_type(16))) __bf16 v16bf;
typedef __attribute__((ext_vector_type(8)))  __bf16 v8bf;
typedef __attribute__((ext_vector_type(8)))  float  v8f;

__device__ __forceinline__ __bf16 f2bf(float f) {
    unsigned u = __builtin_bit_cast(unsigned, f);
    u += 0x7fffu + ((u >> 16) & 1u);           // round-to-nearest-even
    unsigned short hs = (unsigned short)(u >> 16);
    return __builtin_bit_cast(__bf16, hs);
}

// ---------------- h0 = x_emb1[x0] + x_emb2[x1] ----------------
__global__ void h0_kernel(const int* __restrict__ x,
                          const float* __restrict__ emb1,
                          const float* __restrict__ emb2,
                          float* __restrict__ h) {
    long gt = (long)blockIdx.x * blockDim.x + threadIdx.x;
    long node = gt >> 5;
    int lane = (int)(gt & 31);
    if (node >= NNODES) return;
    int i0 = x[node * 2 + 0];
    int i1 = x[node * 2 + 1];
    int d = lane * 4;
    float4 a = *(const float4*)(emb1 + (long)i0 * DIM + d);
    float4 b = *(const float4*)(emb2 + (long)i1 * DIM + d);
    float4 o = make_float4(a.x + b.x, a.y + b.y, a.z + b.z, a.w + b.w);
    *(float4*)(h + node * DIM + d) = o;
}

// ---------------- zero agg + BN stats ----------------
__global__ void zero_kernel(float* __restrict__ agg, float* __restrict__ stats) {
    long idx = (long)blockIdx.x * blockDim.x + threadIdx.x;
    if (idx < (long)NNODES * DIM / 4)
        ((float4*)agg)[idx] = make_float4(0.f, 0.f, 0.f, 0.f);
    if (blockIdx.x == 0 && threadIdx.x < 256)
        stats[threadIdx.x] = 0.f;
}

// ---------------- message gather + scatter-add ----------------
__global__ void scatter_kernel(const float* __restrict__ h,
                               const int* __restrict__ edge_index,  // [2, E]
                               const int* __restrict__ edge_attr,   // [E, 2]
                               const float* __restrict__ e1,        // [6, D]
                               const float* __restrict__ e2,        // [3, D]
                               float* __restrict__ agg) {
    long gt = (long)blockIdx.x * blockDim.x + threadIdx.x;
    long edge = gt >> 5;
    int lane = (int)(gt & 31);
    if (edge >= ETOT) return;
    int r, c, a0, a1;
    if (edge < NEDGES) {
        r  = edge_index[edge];
        c  = edge_index[NEDGES + edge];
        a0 = edge_attr[2 * edge + 0];
        a1 = edge_attr[2 * edge + 1];
    } else {                       // self loop, attr = [4, 0]
        r = c = (int)(edge - NEDGES);
        a0 = 4; a1 = 0;
    }
    int d = lane * 4;
    float4 hv = *(const float4*)(h  + (long)r * DIM + d);
    float4 v1 = *(const float4*)(e1 + (long)a0 * DIM + d);
    float4 v2 = *(const float4*)(e2 + (long)a1 * DIM + d);
    float* dst = agg + (long)c * DIM + d;
    __hip_atomic_fetch_add(dst + 0, hv.x + v1.x + v2.x, __ATOMIC_RELAXED, __HIP_MEMORY_SCOPE_AGENT);
    __hip_atomic_fetch_add(dst + 1, hv.y + v1.y + v2.y, __ATOMIC_RELAXED, __HIP_MEMORY_SCOPE_AGENT);
    __hip_atomic_fetch_add(dst + 2, hv.z + v1.z + v2.z, __ATOMIC_RELAXED, __HIP_MEMORY_SCOPE_AGENT);
    __hip_atomic_fetch_add(dst + 3, hv.w + v1.w + v2.w, __ATOMIC_RELAXED, __HIP_MEMORY_SCOPE_AGENT);
}

// ---------------- pack W[K,Ncols] f32 -> B-fragment bf16 layout ----------------
// Block (kc, nt) = 32 lanes x 16 bf16 (32B/lane, contiguous). Lane l:
//   n = nt*16 + (l&15); kOff = kc*32 + (l>=16 ? 8 : 0); elem i = W[kOff+i+(i>=8?8:0)][n]
__global__ void prep_w_kernel(const float* __restrict__ W,
                              unsigned short* __restrict__ dst,
                              int Kdim, int Ncols) {
    int gt = blockIdx.x * blockDim.x + threadIdx.x;
    int ntiles = Ncols / 16;
    int nblocks = (Kdim / 32) * ntiles;
    if (gt >= nblocks * 32) return;
    int blk = gt >> 5, lane = gt & 31;
    int kc = blk / ntiles, nt = blk % ntiles;
    int n = nt * 16 + (lane & 15);
    int kOff = kc * 32 + ((lane >> 4) << 3);
    v16bf frag;
    #pragma unroll
    for (int i = 0; i < 16; ++i)
        frag[i] = f2bf(W[(long)(kOff + i + (i >= 8 ? 8 : 0)) * Ncols + n]);
    *(v16bf*)(dst + ((long)blk * 32 + lane) * 16) = frag;
}

// ---------------- WMMA GEMM: C = act(A[rows,K] @ Bp + bias) ----------------
// 256 threads = 8 waves; 64-row tile per block; wave -> (mtile = w&3,
// ntiles [ (w>>2)*NTPW , +NTPW )). Packed B (64KB) staged in LDS once per WG.
// ABF16: A rows are bf16 (no inline convert). CBF16: store C as bf16.
template<int KDIM, int NCOLS, int NTPW, bool RELU, bool ABF16, bool CBF16>
__global__ __launch_bounds__(256) void gemm_kernel(
        const void* __restrict__ Ain,
        const unsigned short* __restrict__ Bp,
        const float* __restrict__ bias,
        void* __restrict__ Cout) {
    constexpr int NT   = NCOLS / 16;
    constexpr int NBLK = (KDIM / 32) * NT;           // fragment blocks (64)
    __shared__ unsigned short sB[NBLK * 512];        // 1024 B per block = 64 KB

    const int tid = threadIdx.x;
    {   // cooperative stage: 256 B per thread
        const uint4* src = (const uint4*)Bp;
        uint4* dst = (uint4*)sB;
        #pragma unroll
        for (int i = 0; i < NBLK / 4; ++i)
            dst[tid + i * 256] = src[tid + i * 256];
    }
    __syncthreads();

    const int wave = tid >> 5;
    const int lane = tid & 31;
    const int mtile = wave & 3;
    const int ntStart = (wave >> 2) * NTPW;
    const long mBase = (long)blockIdx.x * 64 + mtile * 16;
    const int laneHalf = lane >> 4;        // 0 or 1
    const int lmod = lane & 15;
    const long arow = mBase + lmod;

    v8f acc[NTPW] = {};

    #pragma unroll
    for (int kc = 0; kc < KDIM / 32; ++kc) {
        // A fragment: lane(<16): M=lmod, K in [k0,k0+8) U [k0+16,k0+24)
        const int k0 = kc * 32 + laneHalf * 8;
        v16bf af;
        if constexpr (ABF16) {
            const unsigned short* ap = (const unsigned short*)Ain + arow * KDIM + k0;
            v8bf lo = *(const v8bf*)(ap);
            v8bf hi = *(const v8bf*)(ap + 16);
            af = __builtin_shufflevector(lo, hi, 0, 1, 2, 3, 4, 5, 6, 7,
                                                 8, 9, 10, 11, 12, 13, 14, 15);
        } else {
            const float* ap = (const float*)Ain + arow * KDIM + k0;
            float4 a0 = *(const float4*)(ap + 0);
            float4 a1 = *(const float4*)(ap + 4);
            float4 a2 = *(const float4*)(ap + 16);
            float4 a3 = *(const float4*)(ap + 20);
            af[0]  = f2bf(a0.x); af[1]  = f2bf(a0.y); af[2]  = f2bf(a0.z); af[3]  = f2bf(a0.w);
            af[4]  = f2bf(a1.x); af[5]  = f2bf(a1.y); af[6]  = f2bf(a1.z); af[7]  = f2bf(a1.w);
            af[8]  = f2bf(a2.x); af[9]  = f2bf(a2.y); af[10] = f2bf(a2.z); af[11] = f2bf(a2.w);
            af[12] = f2bf(a3.x); af[13] = f2bf(a3.y); af[14] = f2bf(a3.z); af[15] = f2bf(a3.w);
        }
        #pragma unroll
        for (int j = 0; j < NTPW; ++j) {
            const int nt = ntStart + j;
            v16bf bf = *(const v16bf*)(sB + ((kc * NT + nt) * 32 + lane) * 16);
            acc[j] = __builtin_amdgcn_wmma_f32_16x16x32_bf16(
                         false, af, false, bf, (short)0, acc[j], false, false);
        }
    }

    // Epilogue: C/D layout — VGPR r: lanes0-15 M=r, lanes16-31 M=8+r; N=lmod
    #pragma unroll
    for (int j = 0; j < NTPW; ++j) {
        const int n = (ntStart + j) * 16 + lmod;
        const float bv = bias[n];
        const long mOff = mBase + laneHalf * 8;
        #pragma unroll
        for (int r = 0; r < 8; ++r) {
            float v = acc[j][r] + bv;
            if (RELU) v = v > 0.f ? v : 0.f;
            if constexpr (CBF16)
                ((unsigned short*)Cout)[(mOff + r) * NCOLS + n] =
                    __builtin_bit_cast(unsigned short, f2bf(v));
            else
                ((float*)Cout)[(mOff + r) * NCOLS + n] = v;
        }
    }
}

// ---------------- BatchNorm pass 1: per-feature sum / sumsq ----------------
__global__ void bn_reduce_kernel(const float* __restrict__ hm,
                                 float* __restrict__ stats) {
    const int f = threadIdx.x & 127;
    const int sub = threadIdx.x >> 7;                // 0..1
    const long rowBase = (long)blockIdx.x * 1024;    // 128 blocks * 1024 rows = N
    float s = 0.f, sq = 0.f;
    for (int i = sub; i < 1024; i += 2) {
        float v = hm[(rowBase + i) * DIM + f];
        s += v; sq += v * v;
    }
    __hip_atomic_fetch_add(&stats[f],       s,  __ATOMIC_RELAXED, __HIP_MEMORY_SCOPE_AGENT);
    __hip_atomic_fetch_add(&stats[128 + f], sq, __ATOMIC_RELAXED, __HIP_MEMORY_SCOPE_AGENT);
}

// ---------------- BatchNorm pass 2: normalize (+optional relu) ----------------
__global__ void bn_apply_kernel(const float* __restrict__ hm,
                                const float* __restrict__ stats,
                                const float* __restrict__ gamma,
                                const float* __restrict__ beta,
                                float* __restrict__ out, int relu) {
    long idx = (long)blockIdx.x * blockDim.x + threadIdx.x;
    if (idx >= (long)NNODES * DIM) return;
    int f = (int)(idx & 127);
    const float invN = 1.f / (float)NNODES;
    float mean = stats[f] * invN;
    float var  = stats[128 + f] * invN - mean * mean;
    float inv  = rsqrtf(var + BN_EPS);
    float v = (hm[idx] - mean) * inv * gamma[f] + beta[f];
    if (relu) v = v > 0.f ? v : 0.f;
    out[idx] = v;
}

extern "C" void kernel_launch(void* const* d_in, const int* in_sizes, int n_in,
                              void* d_out, int out_size, void* d_ws, size_t ws_size,
                              hipStream_t stream) {
    (void)in_sizes; (void)n_in; (void)out_size; (void)ws_size;
    const int*   x          = (const int*)d_in[0];
    const int*   edge_index = (const int*)d_in[1];
    const int*   edge_attr  = (const int*)d_in[2];
    const float* x_emb1     = (const float*)d_in[3];
    const float* x_emb2     = (const float*)d_in[4];
    const float* e_emb1     = (const float*)d_in[5];   // [5,6,128]
    const float* e_emb2     = (const float*)d_in[6];   // [5,3,128]
    const float* W1         = (const float*)d_in[7];   // [5,128,256]
    const float* b1         = (const float*)d_in[8];   // [5,256]
    const float* W2         = (const float*)d_in[9];   // [5,256,128]
    const float* b2         = (const float*)d_in[10];  // [5,128]
    const float* gamma      = (const float*)d_in[11];  // [5,128]
    const float* beta       = (const float*)d_in[12];  // [5,128]
    float* out = (float*)d_out;

    // workspace layout (all offsets 32B-aligned)
    float* h   = (float*)d_ws;                               // N*D f32
    float* agg = h + (size_t)NNODES * DIM;                   // N*D f32 (also hm)
    unsigned short* tbf = (unsigned short*)(agg + (size_t)NNODES * DIM); // N*2D bf16
    float* stats = (float*)(tbf + (size_t)NNODES * 2 * DIM); // 256 f32
    unsigned short* w1p = (unsigned short*)(stats + 256);    // 64 blocks * 512 u16
    unsigned short* w2p = w1p + 64 * 512;                    // 64 blocks * 512 u16

    h0_kernel<<<(NNODES * 32) / 256, 256, 0, stream>>>(x, x_emb1, x_emb2, h);

    for (int l = 0; l < 5; ++l) {
        prep_w_kernel<<<8, 256, 0, stream>>>(W1 + (long)l * 128 * 256, w1p, 128, 256);
        prep_w_kernel<<<8, 256, 0, stream>>>(W2 + (long)l * 256 * 128, w2p, 256, 128);
        zero_kernel<<<((long)NNODES * DIM / 4) / 256, 256, 0, stream>>>(agg, stats);
        scatter_kernel<<<((long)ETOT * 32) / 256, 256, 0, stream>>>(
            h, edge_index, edge_attr,
            e_emb1 + (long)l * 6 * DIM, e_emb2 + (long)l * 3 * DIM, agg);
        gemm_kernel<128, 256, 8, true,  false, true ><<<NNODES / 64, 256, 0, stream>>>(
            agg, w1p, b1 + l * 256, tbf);
        gemm_kernel<256, 128, 4, false, true,  false><<<NNODES / 64, 256, 0, stream>>>(
            tbf, w2p, b2 + l * 128, agg);
        bn_reduce_kernel<<<128, 256, 0, stream>>>(agg, stats);
        bn_apply_kernel<<<((long)NNODES * DIM) / 256, 256, 0, stream>>>(
            agg, stats, gamma + l * 128, beta + l * 128,
            (l == 4) ? out : h, (l < 4) ? 1 : 0);
    }
}
